// SlotDecoder_37881611550791
// MI455X (gfx1250) — compile-verified, matrix operand order checked by reading, not run on hardware
//
#include <hip/hip_runtime.h>
#include <hip/hip_bf16.h>

typedef __bf16 v16bf __attribute__((ext_vector_type(16)));
typedef float v8f __attribute__((ext_vector_type(8)));
typedef float v2f __attribute__((ext_vector_type(2)));
typedef unsigned int ui8v __attribute__((ext_vector_type(8)));

__device__ __forceinline__ unsigned short bf16r(float x) {
  unsigned u = __float_as_uint(x);
  u += 0x7FFFu + ((u >> 16) & 1u);
  return (unsigned short)(u >> 16);
}

__device__ __forceinline__ v8f zero8() {
  v8f z;
  for (int i = 0; i < 8; ++i) z[i] = 0.0f;
  return z;
}

// ---------------------------------------------------------------------------
// Kernel A: fused LayerNorm(features) -> bf16, K = x@Wk, V = x@Wv  (WMMA bf16)
// 128 rows per block, 8 waves, each wave owns a 16-row slice and both 16x64
// output tiles (K and V), accumulating over 24 K-chunks of 32.
// ---------------------------------------------------------------------------
__global__ __launch_bounds__(256) void ln_kv_kernel(
    const float* __restrict__ features, const float* __restrict__ g,
    const float* __restrict__ beta, const float* __restrict__ Wk,
    const float* __restrict__ Wv, float* __restrict__ kout,
    float* __restrict__ vout) {
  __shared__ float ps[256], ps2[256];
  __shared__ float smu[128], srs[128];
  __shared__ unsigned short sx[128 * 32];   // normalized x chunk, bf16
  __shared__ unsigned short swk[64 * 32];   // Wk chunk, transposed [n][k], bf16
  __shared__ unsigned short swv[64 * 32];   // Wv chunk, transposed [n][k], bf16

  const int tid = threadIdx.x;
  const int row0 = blockIdx.x * 128;

  // ---- pass 1: per-row mean/var (2 threads per row) ----
  {
    int r = tid >> 1, h = tid & 1;
    const float* p = features + (size_t)(row0 + r) * 768 + h * 384;
    float s = 0.f, s2 = 0.f;
    for (int i = 0; i < 384; ++i) { float x = p[i]; s += x; s2 += x * x; }
    ps[tid] = s; ps2[tid] = s2;
  }
  __syncthreads();
  if (tid < 128) {
    float s = ps[2 * tid] + ps[2 * tid + 1];
    float s2 = ps2[2 * tid] + ps2[2 * tid + 1];
    float mu = s * (1.0f / 768.0f);
    float var = s2 * (1.0f / 768.0f) - mu * mu;
    smu[tid] = mu;
    srs[tid] = rsqrtf(var + 1e-5f);
  }
  __syncthreads();

  const int w = tid >> 5, lane = tid & 31;
  const int m = lane & 15, hi = lane >> 4;

  v8f accK[4], accV[4];
  for (int t = 0; t < 4; ++t) { accK[t] = zero8(); accV[t] = zero8(); }

  // per-lane K-pair base offsets for the 16-bit A/B fragment layout
  int K0[8];
  for (int p = 0; p < 8; ++p) K0[p] = 2 * p + (p >= 4 ? 8 : 0) + 8 * hi;

  for (int kk = 0; kk < 768; kk += 32) {
    // normalized x chunk -> bf16 LDS (coalesced over columns)
    for (int j = 0; j < 16; ++j) {
      int id = tid + j * 256;           // 4096 = 128*32
      int r = id >> 5, c = id & 31;
      float x = features[(size_t)(row0 + r) * 768 + kk + c];
      x = (x - smu[r]) * srs[r] * g[kk + c] + beta[kk + c];
      sx[r * 32 + c] = bf16r(x);
    }
    // weight chunks, transposed to [n][k] (coalesced global reads over n)
    for (int j = 0; j < 8; ++j) {
      int id = tid + j * 256;           // 2048 = 32*64
      int c = id >> 6, n = id & 63;
      swk[n * 32 + c] = bf16r(Wk[(size_t)(kk + c) * 64 + n]);
      swv[n * 32 + c] = bf16r(Wv[(size_t)(kk + c) * 64 + n]);
    }
    __syncthreads();

    // A fragment: 16x32 bf16, rows 16*w .. 16*w+15
    ui8v au;
    const unsigned short* xrow = &sx[(16 * w + m) * 32];
    for (int p = 0; p < 8; ++p) au[p] = *(const unsigned*)(xrow + K0[p]);
    v16bf a = __builtin_bit_cast(v16bf, au);

    for (int t = 0; t < 4; ++t) {
      ui8v bu;
      const unsigned short* wrowk = &swk[(16 * t + m) * 32];
      for (int p = 0; p < 8; ++p) bu[p] = *(const unsigned*)(wrowk + K0[p]);
      v16bf bk = __builtin_bit_cast(v16bf, bu);
      accK[t] = __builtin_amdgcn_wmma_f32_16x16x32_bf16(
          false, a, false, bk, (short)0, accK[t], false, false);

      const unsigned short* wrowv = &swv[(16 * t + m) * 32];
      for (int p = 0; p < 8; ++p) bu[p] = *(const unsigned*)(wrowv + K0[p]);
      v16bf bv = __builtin_bit_cast(v16bf, bu);
      accV[t] = __builtin_amdgcn_wmma_f32_16x16x32_bf16(
          false, a, false, bv, (short)0, accV[t], false, false);
    }
    __syncthreads();
  }

  // C/D layout: vgpr r -> M = r (lanes 0-15) / M = r+8 (lanes 16-31)
  for (int t = 0; t < 4; ++t)
    for (int r = 0; r < 8; ++r) {
      size_t row = (size_t)(row0 + 16 * w + r + 8 * hi);
      kout[row * 64 + 16 * t + m] = accK[t][r];
      vout[row * 64 + 16 * t + m] = accV[t][r];
    }
}

// ---------------------------------------------------------------------------
// Kernel B: 3 slot-attention iterations, one block per batch element.
// k/v cached in LDS (padded to 208 rows), attention einsums on WMMA f32
// 16x16x4, softmax/GRU/MLP on VALU.
// ---------------------------------------------------------------------------
__global__ __launch_bounds__(256) void slot_iter_kernel(
    const float* __restrict__ kws, const float* __restrict__ vws,
    const float* __restrict__ slots_init, const float* __restrict__ ln_s_g,
    const float* __restrict__ ln_s_b, const float* __restrict__ Wq,
    const float* __restrict__ W_ih, const float* __restrict__ W_hh,
    const float* __restrict__ b_ih, const float* __restrict__ b_hh,
    const float* __restrict__ ln_m_g, const float* __restrict__ ln_m_b,
    const float* __restrict__ W1, const float* __restrict__ b1,
    const float* __restrict__ W2, const float* __restrict__ b2,
    float* __restrict__ slots_out) {
  __shared__ float sk[208 * 64];
  __shared__ float sv[208 * 64];
  __shared__ float sdots[16 * 208];
  __shared__ float sq[16 * 64];
  __shared__ float sxs[640];
  __shared__ float sslots[640];
  __shared__ float sprev[640];
  __shared__ float spart[8 * 256];
  __shared__ float supd[640];
  __shared__ float sgx[1920];
  __shared__ float sgh[1920];
  __shared__ float shh[1280];
  __shared__ float smu[16], srs[16], srsum[16];

  const int tid = threadIdx.x;
  const int b = blockIdx.x;
  const int w = tid >> 5, lane = tid & 31;
  const int m = lane & 15, hi = lane >> 4;

  // load k, v (zero-pad rows 197..207 so WMMA tiles read zeros)
  for (int idx = tid; idx < 208 * 64; idx += 256) {
    int r = idx >> 6, c = idx & 63;
    float kv = 0.f, vv = 0.f;
    if (r < 197) {
      size_t gr = ((size_t)b * 197 + r) * 64 + c;
      kv = kws[gr]; vv = vws[gr];
    }
    sk[idx] = kv; sv[idx] = vv;
  }
  for (int idx = tid; idx < 640; idx += 256) sslots[idx] = slots_init[idx];
  __syncthreads();

  for (int it = 0; it < 3; ++it) {
    for (int idx = tid; idx < 640; idx += 256) sprev[idx] = sslots[idx];
    __syncthreads();

    // LN(slots) -> sxs
    if (tid < 10) {
      float s = 0.f, s2 = 0.f;
      for (int d = 0; d < 64; ++d) { float x = sslots[tid * 64 + d]; s += x; s2 += x * x; }
      float mu = s * (1.f / 64.f);
      float var = s2 * (1.f / 64.f) - mu * mu;
      smu[tid] = mu; srs[tid] = rsqrtf(var + 1e-5f);
    }
    __syncthreads();
    for (int idx = tid; idx < 640; idx += 256) {
      int s = idx >> 6, d = idx & 63;
      sxs[idx] = (sslots[idx] - smu[s]) * srs[s] * ln_s_g[d] + ln_s_b[d];
    }
    __syncthreads();

    // q = LN(slots) @ Wq  (rows 10..15 zero-padded)
    for (int idx = tid; idx < 1024; idx += 256) {
      int s = idx >> 6, d = idx & 63;
      float acc = 0.f;
      if (s < 10)
        for (int kq = 0; kq < 64; ++kq) acc += sxs[s * 64 + kq] * Wq[kq * 64 + d];
      sq[idx] = acc;
    }
    __syncthreads();

    // dots = scale * q @ k^T  -> sdots[16][208]  (WMMA f32 16x16x4)
    for (int t = w; t < 13; t += 8) {
      v8f acc = zero8();
      int n0 = 16 * t;
      for (int kc = 0; kc < 16; ++kc) {
        int kb = kc * 4 + 2 * hi;
        v2f a, bb;
        a[0] = sq[m * 64 + kb];           a[1] = sq[m * 64 + kb + 1];
        bb[0] = sk[(n0 + m) * 64 + kb];   bb[1] = sk[(n0 + m) * 64 + kb + 1];
        acc = __builtin_amdgcn_wmma_f32_16x16x4_f32(
            false, a, false, bb, (short)0, acc, false, false);
      }
      for (int r = 0; r < 8; ++r)
        sdots[(r + 8 * hi) * 208 + n0 + m] = acc[r] * 0.125f;  // D^-0.5
    }
    __syncthreads();

    // softmax over slots (axis=1) per input column, + eps
    for (int n = tid; n < 197; n += 256) {
      float mx = -1e30f;
      for (int s = 0; s < 10; ++s) mx = fmaxf(mx, sdots[s * 208 + n]);
      float e[10], sum = 0.f;
      for (int s = 0; s < 10; ++s) { e[s] = expf(sdots[s * 208 + n] - mx); sum += e[s]; }
      float inv = 1.f / sum;
      for (int s = 0; s < 10; ++s) sdots[s * 208 + n] = e[s] * inv + 1e-8f;
    }
    __syncthreads();
    if (tid < 10) {
      float s = 0.f;
      for (int n = 0; n < 197; ++n) s += sdots[tid * 208 + n];
      srsum[tid] = 1.f / s;
    }
    __syncthreads();
    for (int idx = tid; idx < 10 * 197; idx += 256) {
      int s = idx / 197, n = idx % 197;
      sdots[s * 208 + n] *= srsum[s];
    }
    __syncthreads();

    // updates = attn @ v  (WMMA f32, K split over wave halves, LDS reduce)
    {
      int n0 = (w & 3) * 16;
      int c0 = (w >> 2) * 26;               // 52 K-chunks total
      v8f acc = zero8();
      for (int kc = c0; kc < c0 + 26; ++kc) {
        int kb = kc * 4 + 2 * hi;
        v2f a, bb;
        a[0] = sdots[m * 208 + kb];         a[1] = sdots[m * 208 + kb + 1];
        bb[0] = sv[kb * 64 + n0 + m];       bb[1] = sv[(kb + 1) * 64 + n0 + m];
        acc = __builtin_amdgcn_wmma_f32_16x16x4_f32(
            false, a, false, bb, (short)0, acc, false, false);
      }
      for (int r = 0; r < 8; ++r)
        spart[w * 256 + (r + 8 * hi) * 16 + m] = acc[r];
    }
    __syncthreads();
    for (int idx = tid; idx < 640; idx += 256) {
      int s = idx >> 6, d = idx & 63;
      int t4 = d >> 4, dd = d & 15;
      supd[idx] = spart[t4 * 256 + s * 16 + dd] + spart[(t4 + 4) * 256 + s * 16 + dd];
    }
    __syncthreads();

    // GRU gates: gx = upd @ W_ih^T + b_ih ; gh = prev @ W_hh^T + b_hh
    for (int idx = tid; idx < 1920; idx += 256) {
      int s = idx / 192, j = idx % 192;
      float ax = b_ih[j], ah = b_hh[j];
      const float* wi = &W_ih[j * 64];
      const float* wh = &W_hh[j * 64];
      const float* u = &supd[s * 64];
      const float* pv = &sprev[s * 64];
      for (int d = 0; d < 64; ++d) { ax += u[d] * wi[d]; ah += pv[d] * wh[d]; }
      sgx[idx] = ax; sgh[idx] = ah;
    }
    __syncthreads();
    for (int idx = tid; idx < 640; idx += 256) {
      int s = idx >> 6, d = idx & 63;
      float r = 1.f / (1.f + expf(-(sgx[s * 192 + d] + sgh[s * 192 + d])));
      float z = 1.f / (1.f + expf(-(sgx[s * 192 + 64 + d] + sgh[s * 192 + 64 + d])));
      float nn = tanhf(sgx[s * 192 + 128 + d] + r * sgh[s * 192 + 128 + d]);
      sslots[idx] = (1.f - z) * nn + z * sprev[idx];
    }
    __syncthreads();

    // residual MLP: slots += relu(LN(slots)@W1 + b1)@W2 + b2
    if (tid < 10) {
      float s = 0.f, s2 = 0.f;
      for (int d = 0; d < 64; ++d) { float x = sslots[tid * 64 + d]; s += x; s2 += x * x; }
      float mu = s * (1.f / 64.f);
      float var = s2 * (1.f / 64.f) - mu * mu;
      smu[tid] = mu; srs[tid] = rsqrtf(var + 1e-5f);
    }
    __syncthreads();
    for (int idx = tid; idx < 640; idx += 256) {
      int s = idx >> 6, d = idx & 63;
      sxs[idx] = (sslots[idx] - smu[s]) * srs[s] * ln_m_g[d] + ln_m_b[d];
    }
    __syncthreads();
    for (int idx = tid; idx < 1280; idx += 256) {
      int s = idx >> 7, j = idx & 127;
      float acc = b1[j];
      for (int d = 0; d < 64; ++d) acc += sxs[s * 64 + d] * W1[d * 128 + j];
      shh[idx] = fmaxf(acc, 0.f);
    }
    __syncthreads();
    for (int idx = tid; idx < 640; idx += 256) {
      int s = idx >> 6, d = idx & 63;
      float acc = b2[d];
      for (int j = 0; j < 128; ++j) acc += shh[s * 128 + j] * W2[j * 64 + d];
      sslots[idx] += acc;
    }
    __syncthreads();
  }

  for (int idx = tid; idx < 640; idx += 256)
    slots_out[(size_t)b * 640 + idx] = sslots[idx];
}

// ---------------------------------------------------------------------------
// Kernel C: out = relu(slots @ Wo + bo)   [2560x64]x[64x768], WMMA f32
// One 16x16 tile per wave; 160x48 tiles; 8 waves/block -> 960 blocks.
// ---------------------------------------------------------------------------
__global__ __launch_bounds__(256) void out_proj_kernel(
    const float* __restrict__ slots, const float* __restrict__ Wo,
    const float* __restrict__ bo, float* __restrict__ out) {
  const int tid = threadIdx.x;
  const int w = tid >> 5, lane = tid & 31;
  const int m = lane & 15, hi = lane >> 4;
  int gw = blockIdx.x * 8 + w;          // 0..7679
  int tm = gw / 48, tn = gw % 48;
  int m0 = tm * 16, n0 = tn * 16;

  v8f acc = zero8();
  for (int kc = 0; kc < 16; ++kc) {
    int kb = kc * 4 + 2 * hi;
    v2f a, bb;
    a[0] = slots[(size_t)(m0 + m) * 64 + kb];
    a[1] = slots[(size_t)(m0 + m) * 64 + kb + 1];
    bb[0] = Wo[(size_t)kb * 768 + n0 + m];
    bb[1] = Wo[(size_t)(kb + 1) * 768 + n0 + m];
    acc = __builtin_amdgcn_wmma_f32_16x16x4_f32(
        false, a, false, bb, (short)0, acc, false, false);
  }
  float bias = bo[n0 + m];
  for (int r = 0; r < 8; ++r)
    out[(size_t)(m0 + r + 8 * hi) * 768 + n0 + m] = fmaxf(acc[r] + bias, 0.f);
}

// ---------------------------------------------------------------------------
extern "C" void kernel_launch(void* const* d_in, const int* in_sizes, int n_in,
                              void* d_out, int out_size, void* d_ws, size_t ws_size,
                              hipStream_t stream) {
  const float* features  = (const float*)d_in[0];
  const float* ln_in_g   = (const float*)d_in[1];
  const float* ln_in_b   = (const float*)d_in[2];
  const float* slots_init= (const float*)d_in[3];
  const float* ln_s_g    = (const float*)d_in[4];
  const float* ln_s_b    = (const float*)d_in[5];
  const float* Wq        = (const float*)d_in[6];
  const float* Wk        = (const float*)d_in[7];
  const float* Wv        = (const float*)d_in[8];
  const float* W_ih      = (const float*)d_in[9];
  const float* W_hh      = (const float*)d_in[10];
  const float* b_ih      = (const float*)d_in[11];
  const float* b_hh      = (const float*)d_in[12];
  const float* ln_m_g    = (const float*)d_in[13];
  const float* ln_m_b    = (const float*)d_in[14];
  const float* W1        = (const float*)d_in[15];
  const float* b1        = (const float*)d_in[16];
  const float* W2        = (const float*)d_in[17];
  const float* b2        = (const float*)d_in[18];
  const float* Wo        = (const float*)d_in[19];
  const float* bo        = (const float*)d_in[20];

  float* outp = (float*)d_out;
  float* slots_out = outp + (size_t)256 * 10 * 768;   // tuple: (out, slots)
  float* kws = (float*)d_ws;                          // [50432, 64]
  float* vws = kws + (size_t)50432 * 64;              // [50432, 64]

  ln_kv_kernel<<<394, 256, 0, stream>>>(features, ln_in_g, ln_in_b, Wk, Wv, kws, vws);
  slot_iter_kernel<<<256, 256, 0, stream>>>(kws, vws, slots_init, ln_s_g, ln_s_b, Wq,
                                            W_ih, W_hh, b_ih, b_hh, ln_m_g, ln_m_b,
                                            W1, b1, W2, b2, slots_out);
  out_proj_kernel<<<960, 256, 0, stream>>>(slots_out, Wo, bo, outp);
}
